// EncoderHR_88029649699181
// MI455X (gfx1250) — matrix operand order repaired and checked by reading, not compiled, not run on hardware
//
#include <hip/hip_runtime.h>
#include <hip/hip_bf16.h>

typedef __attribute__((ext_vector_type(16))) __bf16 v16bf;
typedef __attribute__((ext_vector_type(8)))  __bf16 v8bf;
typedef __attribute__((ext_vector_type(8)))  float  v8f;

#define DIRS   2
#define BATCH  128
#define SEQ    256
#define EIN    825
#define EPAD   832     // 26 * 32
#define HID    600
#define HPAD   608     // 19 * 32
#define NGATE  2400    // 4 * HID, 150 N-tiles
#define NBLK   64      // 16 (d,mt) pairs x 4 sub-blocks
#define NTPW   5       // N-tiles per wave in LSTM phase A

// CDNA5 async global->LDS copies (ASYNCcnt path), when the toolchain declares them.
// Param types per hipcc diagnostic: (v4i AS1*, v4i AS3*, imm offset, imm cpol).
#if defined(__AMDGCN__) && __has_builtin(__builtin_amdgcn_global_load_async_to_lds_b128)
#define USE_ASYNC_LDS 1
typedef int v4i_ __attribute__((vector_size(4 * sizeof(int))));
typedef __attribute__((address_space(1))) v4i_ as1_v4i;   // global
typedef __attribute__((address_space(3))) v4i_ as3_v4i;   // LDS
#endif

__device__ __forceinline__ void async_wait0() {
#if defined(__AMDGCN__)
#if __has_builtin(__builtin_amdgcn_s_wait_asynccnt)
  __builtin_amdgcn_s_wait_asynccnt(0);
#elif defined(USE_ASYNC_LDS)
  asm volatile("s_wait_asynccnt 0x0" ::: "memory");
#endif
#endif
}

union BfFrag { v16bf v; v8bf h[2]; };

// A-matrix fragment (16x32 bf16, wave32): lane group g = lane>>4, row = lane&15.
// Lane holds 16 contiguous K values starting at kbase+16g (matches B).
__device__ __forceinline__ v16bf load_a_frag(const __bf16* row, int kbase, int g) {
  BfFrag u;
  u.h[0] = *(const v8bf*)(row + kbase + 16 * g);
  u.h[1] = *(const v8bf*)(row + kbase + 16 * g + 8);
  return u.v;
}
// B-matrix fragment (32x16 bf16): lane = column n (lane&15), group g = lane>>4.
// Lane holds 16 contiguous K values starting at kbase+16g.
__device__ __forceinline__ v16bf load_b_frag(const __bf16* row, int kbase, int g) {
  return *(const v16bf*)(row + kbase + 16 * g);
}

__device__ __forceinline__ float sigmoidf_(float x) { return 1.0f / (1.0f + __expf(-x)); }

// Device-wide generation barrier (persistent grid, all NBLK blocks resident).
__device__ __forceinline__ void gbar(int* cnt, int* gen) {
  __syncthreads();
  if (threadIdx.x == 0) {
    __threadfence();
    int gv = atomicAdd(gen, 0);
    if (atomicAdd(cnt, 1) == NBLK - 1) {
      atomicExch(cnt, 0);
      __threadfence();
      atomicAdd(gen, 1);
    } else {
      while (atomicAdd(gen, 0) == gv) { __builtin_amdgcn_s_sleep(8); }
    }
    __threadfence();
  }
  __syncthreads();
}

// ---------- conversion / init kernels ----------
__global__ __launch_bounds__(256)
void cvt_pad_kernel(const float* __restrict__ src, __bf16* __restrict__ dst,
                    int srows, int scols, int drows, int dcols) {
  int idx = blockIdx.x * 256 + threadIdx.x;
  int total = drows * dcols;
  if (idx >= total) return;
  int rr = idx / dcols;
  int cc = idx - rr * dcols;
  float v = (rr < srows && cc < scols) ? src[(size_t)rr * scols + cc] : 0.0f;
  dst[idx] = (__bf16)v;
}

__global__ __launch_bounds__(256)
void copyf_kernel(const float* __restrict__ src, float* __restrict__ dst, int n) {
  int idx = blockIdx.x * 256 + threadIdx.x;
  if (idx < n) dst[idx] = src[idx];
}

__global__ void zero_ints_kernel(int* p, int n) {
  int idx = blockIdx.x * blockDim.x + threadIdx.x;
  if (idx < n) p[idx] = 0;
}

// ---------- highway layer: y = sig(xWg^T+bg)*relu(xWn^T+bn) + (1-sig)*(xWl^T+bl) ----------
// grid = (2048 M-tiles, 13), block = 128 (4 waves; wave w handles F-tile blockIdx.y*4+w).
// 3 WMMAs per K-step share a single A fragment.
__global__ __launch_bounds__(128)
void highway_kernel(const __bf16* __restrict__ xin, __bf16* __restrict__ xout,
                    const __bf16* __restrict__ Wg, const __bf16* __restrict__ Wn,
                    const __bf16* __restrict__ Wl, const float* __restrict__ bg,
                    const float* __restrict__ bn, const float* __restrict__ bl) {
  const int lane = threadIdx.x & 31;
  const int g = lane >> 4, r = lane & 15;
  const int mt = blockIdx.x;
  const int ft = blockIdx.y * 4 + (threadIdx.x >> 5);

  const __bf16* arow = xin + (size_t)(mt * 16 + r) * EPAD;
  const __bf16* bgr = Wg + (size_t)(ft * 16 + r) * EPAD;
  const __bf16* bnr = Wn + (size_t)(ft * 16 + r) * EPAD;
  const __bf16* blr = Wl + (size_t)(ft * 16 + r) * EPAD;

  v8f ag = {}, an = {}, al = {};
  for (int k = 0; k < EPAD / 32; ++k) {
    __builtin_prefetch(bgr + k * 32 + 512, 0, 1);  // global_prefetch_b8 ahead on weight stream
    v16bf a = load_a_frag(arow, k * 32, g);
    v16bf b = load_b_frag(bgr, k * 32, g);
    ag = __builtin_amdgcn_wmma_f32_16x16x32_bf16(false, a, false, b, (short)0, ag, false, false);
    b = load_b_frag(bnr, k * 32, g);
    an = __builtin_amdgcn_wmma_f32_16x16x32_bf16(false, a, false, b, (short)0, an, false, false);
    b = load_b_frag(blr, k * 32, g);
    al = __builtin_amdgcn_wmma_f32_16x16x32_bf16(false, a, false, b, (short)0, al, false, false);
  }

  const int f = ft * 16 + r;            // D col = lane&15
  const bool valid = (f < EIN);
  const float bgv = valid ? bg[f] : 0.0f;
  const float bnv = valid ? bn[f] : 0.0f;
  const float blv = valid ? bl[f] : 0.0f;
#pragma unroll
  for (int rr = 0; rr < 8; ++rr) {
    int m = mt * 16 + rr + 8 * g;       // D row = vgpr + 8*(lane>>4)
    float gt = sigmoidf_(ag[rr] + bgv);
    float nl = fmaxf(an[rr] + bnv, 0.0f);
    float ln = al[rr] + blv;
    float y = valid ? (gt * nl + (1.0f - gt) * ln) : 0.0f;  // keep K-padding zero
    xout[(size_t)m * EPAD + f] = (__bf16)y;
  }
}

// ---------- persistent bidirectional LSTM scan ----------
// Block -> fixed (dir d, M-tile mt); per step: async-stage x_t and h A-tiles into LDS,
// WMMA GEMM (K = 832 + 608) with A from LDS and weights streamed from L2,
// global barrier, elementwise cell update, barrier.
__global__ __launch_bounds__(256)
void lstm_kernel(const __bf16* __restrict__ xh, const __bf16* __restrict__ Wih,
                 const __bf16* __restrict__ Whh, const float* __restrict__ b_ih,
                 const float* __restrict__ b_hh, __bf16* __restrict__ hbf,
                 float* __restrict__ cst, float* __restrict__ gates,
                 float* __restrict__ out, int* __restrict__ bar) {
  __shared__ __bf16 sx[16 * EPAD];    // x_t A-tile  (26.0 KB)
  __shared__ __bf16 shh[16 * HPAD];   // h   A-tile  (19.0 KB)

  const int lane = threadIdx.x & 31;
  const int g = lane >> 4, r = lane & 15;
  const int wave = threadIdx.x >> 5;
  const int pair = blockIdx.x >> 2;   // 0..15 : (d, mt)
  const int sub  = blockIdx.x & 3;    // 0..3  : N-tile phase
  const int d    = pair >> 3;
  const int mt   = pair & 7;
  int* cnt = bar;
  int* gen = bar + 1;

  // This wave's N-tiles and weight-row pointers (nt = sub + 4*(wave + 8*i)).
  int  ntv[NTPW];
  bool ok[NTPW];
  const __bf16* browE[NTPW];
  const __bf16* browH[NTPW];
#pragma unroll
  for (int i = 0; i < NTPW; ++i) {
    int nt = sub + 4 * (wave + 8 * i);
    ok[i] = (nt < 150);
    ntv[i] = ok[i] ? nt : 0;
    browE[i] = Wih + ((size_t)d * NGATE + ntv[i] * 16 + r) * EPAD;
    browH[i] = Whh + ((size_t)d * NGATE + ntv[i] * 16 + r) * HPAD;
  }

  const int XCH = 16 * (EPAD * 2 / 16);   // 1664 16B chunks of x_t tile
  const int HCH = 16 * (HPAD * 2 / 16);   // 1216 16B chunks of h tile

  for (int t = 0; t < SEQ; ++t) {
    // ---- stage A-tiles into LDS (ASYNCcnt path when available) ----
    for (int i = threadIdx.x; i < XCH + HCH; i += 256) {
      const char* gp;
      char* lp;
      if (i < XCH) {
        int row = i / 104, c = i - row * 104;      // 104 = EPAD*2/16
        gp = (const char*)(xh + ((size_t)(mt * 16 + row) * SEQ + t) * EPAD) + c * 16;
        lp = (char*)(sx + row * EPAD) + c * 16;
      } else {
        int j = i - XCH;
        int row = j / 76, c = j - row * 76;        // 76 = HPAD*2/16
        gp = (const char*)(hbf + ((size_t)d * BATCH + mt * 16 + row) * HPAD) + c * 16;
        lp = (char*)(shh + row * HPAD) + c * 16;
      }
#if defined(USE_ASYNC_LDS)
      __builtin_amdgcn_global_load_async_to_lds_b128((as1_v4i*)gp, (as3_v4i*)lp, 0, 0);
#else
      *(uint4*)lp = *(const uint4*)gp;
#endif
    }
    async_wait0();
    __syncthreads();

    // ---- phase A: gates = x_t@Wih^T + h@Whh^T (A from LDS via ds_load_b128) ----
    v8f acc[NTPW] = {};
    for (int k = 0; k < EPAD / 32; ++k) {
      __builtin_prefetch(browE[0] + k * 32 + 512, 0, 1);
      v16bf a = load_a_frag(sx + r * EPAD, k * 32, g);
#pragma unroll
      for (int i = 0; i < NTPW; ++i) {
        v16bf b = load_b_frag(browE[i], k * 32, g);
        acc[i] = __builtin_amdgcn_wmma_f32_16x16x32_bf16(false, a, false, b, (short)0, acc[i], false, false);
      }
    }
    for (int k = 0; k < HPAD / 32; ++k) {
      v16bf a = load_a_frag(shh + r * HPAD, k * 32, g);
#pragma unroll
      for (int i = 0; i < NTPW; ++i) {
        v16bf b = load_b_frag(browH[i], k * 32, g);
        acc[i] = __builtin_amdgcn_wmma_f32_16x16x32_bf16(false, a, false, b, (short)0, acc[i], false, false);
      }
    }
#pragma unroll
    for (int i = 0; i < NTPW; ++i) {
      if (!ok[i]) continue;
      float* gout = gates + (size_t)d * BATCH * NGATE + ntv[i] * 16 + r;
#pragma unroll
      for (int rr = 0; rr < 8; ++rr) {
        int m = mt * 16 + rr + 8 * g;
        gout[(size_t)m * NGATE] = acc[i][rr];
      }
    }
    gbar(cnt, gen);

    // ---- phase B: LSTM cell update over 2*128*600 elements ----
    for (int idx = blockIdx.x * 256 + threadIdx.x; idx < DIRS * BATCH * HID; idx += NBLK * 256) {
      const int dd = idx / (BATCH * HID);
      const int rm = idx - dd * BATCH * HID;
      const int b = rm / HID;
      const int hh = rm - b * HID;
      const float* grow = gates + ((size_t)dd * BATCH + b) * NGATE;
      const float* bi = b_ih + dd * NGATE;
      const float* bh = b_hh + dd * NGATE;
      float iv = grow[hh] + bi[hh] + bh[hh];
      float fv = grow[HID + hh] + bi[HID + hh] + bh[HID + hh];
      float gv = grow[2 * HID + hh] + bi[2 * HID + hh] + bh[2 * HID + hh];
      float ov = grow[3 * HID + hh] + bi[3 * HID + hh] + bh[3 * HID + hh];
      size_t ci = (size_t)(dd * BATCH + b) * HID + hh;
      float cold = cst[ci];
      float cnew = sigmoidf_(fv) * cold + sigmoidf_(iv) * tanhf(gv);
      float hnew = sigmoidf_(ov) * tanhf(cnew);
      cst[ci] = cnew;
      hbf[((size_t)dd * BATCH + b) * HPAD + hh] = (__bf16)hnew;  // padded cols stay 0
      // context[t][b][dd*H + hh] : streaming output, bypass-cache store
      __builtin_nontemporal_store(hnew, &out[614400 + ((size_t)t * BATCH + b) * (2 * HID) + dd * HID + hh]);
      if (t == SEQ - 1) {
        out[(size_t)b * (2 * HID) + dd * HID + hh] = hnew;            // encoder_outputs
        out[153600 + (size_t)b * (2 * HID) + dd * HID + hh] = hnew;   // final_state
        out[307200 + ci] = hnew;                                      // h_n
        out[460800 + ci] = cnew;                                      // c_n
      }
    }
    gbar(cnt, gen);
  }
}

extern "C" void kernel_launch(void* const* d_in, const int* in_sizes, int n_in,
                              void* d_out, int out_size, void* d_ws, size_t ws_size,
                              hipStream_t stream) {
  (void)in_sizes; (void)n_in; (void)out_size; (void)ws_size;
  const float* x    = (const float*)d_in[0];
  const float* h0   = (const float*)d_in[1];
  const float* c0   = (const float*)d_in[2];
  const float* hwWg = (const float*)d_in[3];
  const float* hwbg = (const float*)d_in[4];
  const float* hwWn = (const float*)d_in[5];
  const float* hwbn = (const float*)d_in[6];
  const float* hwWl = (const float*)d_in[7];
  const float* hwbl = (const float*)d_in[8];
  const float* Wih  = (const float*)d_in[9];
  const float* Whh  = (const float*)d_in[10];
  const float* bih  = (const float*)d_in[11];
  const float* bhh  = (const float*)d_in[12];
  float* out = (float*)d_out;

  char* ws = (char*)d_ws;
  size_t off = 0;
  auto alloc = [&](size_t bytes) -> void* {
    void* p = ws + off;
    off = (off + bytes + 255) & ~(size_t)255;
    return p;
  };
  __bf16* XH0  = (__bf16*)alloc((size_t)BATCH * SEQ * EPAD * 2);
  __bf16* XH1  = (__bf16*)alloc((size_t)BATCH * SEQ * EPAD * 2);
  __bf16* WGb  = (__bf16*)alloc((size_t)2 * EPAD * EPAD * 2);
  __bf16* WNb  = (__bf16*)alloc((size_t)2 * EPAD * EPAD * 2);
  __bf16* WLb  = (__bf16*)alloc((size_t)2 * EPAD * EPAD * 2);
  __bf16* WIHb = (__bf16*)alloc((size_t)DIRS * NGATE * EPAD * 2);
  __bf16* WHHb = (__bf16*)alloc((size_t)DIRS * NGATE * HPAD * 2);
  __bf16* HBF  = (__bf16*)alloc((size_t)DIRS * BATCH * HPAD * 2);
  float*  CST  = (float*)alloc((size_t)DIRS * BATCH * HID * 4);
  float*  GT   = (float*)alloc((size_t)DIRS * BATCH * NGATE * 4);
  int*    BAR  = (int*)alloc(256);

  auto nblocks = [](size_t n) { return (unsigned)((n + 255) / 256); };

  // --- convert + zero-pad to bf16 ---
  cvt_pad_kernel<<<nblocks((size_t)BATCH * SEQ * EPAD), 256, 0, stream>>>(
      x, XH0, BATCH * SEQ, EIN, BATCH * SEQ, EPAD);
  for (int l = 0; l < 2; ++l) {
    cvt_pad_kernel<<<nblocks((size_t)EPAD * EPAD), 256, 0, stream>>>(
        hwWg + (size_t)l * EIN * EIN, WGb + (size_t)l * EPAD * EPAD, EIN, EIN, EPAD, EPAD);
    cvt_pad_kernel<<<nblocks((size_t)EPAD * EPAD), 256, 0, stream>>>(
        hwWn + (size_t)l * EIN * EIN, WNb + (size_t)l * EPAD * EPAD, EIN, EIN, EPAD, EPAD);
    cvt_pad_kernel<<<nblocks((size_t)EPAD * EPAD), 256, 0, stream>>>(
        hwWl + (size_t)l * EIN * EIN, WLb + (size_t)l * EPAD * EPAD, EIN, EIN, EPAD, EPAD);
  }
  for (int d = 0; d < DIRS; ++d) {
    cvt_pad_kernel<<<nblocks((size_t)NGATE * EPAD), 256, 0, stream>>>(
        Wih + (size_t)d * NGATE * EIN, WIHb + (size_t)d * NGATE * EPAD, NGATE, EIN, NGATE, EPAD);
    cvt_pad_kernel<<<nblocks((size_t)NGATE * HPAD), 256, 0, stream>>>(
        Whh + (size_t)d * NGATE * HID, WHHb + (size_t)d * NGATE * HPAD, NGATE, HID, NGATE, HPAD);
    cvt_pad_kernel<<<nblocks((size_t)BATCH * HPAD), 256, 0, stream>>>(
        h0 + (size_t)d * BATCH * HID, HBF + (size_t)d * BATCH * HPAD, BATCH, HID, BATCH, HPAD);
  }
  copyf_kernel<<<nblocks(DIRS * BATCH * HID), 256, 0, stream>>>(c0, CST, DIRS * BATCH * HID);
  zero_ints_kernel<<<1, 64, 0, stream>>>(BAR, 64);

  // --- highway stack (2 layers, ping-pong) ---
  dim3 hgrid((BATCH * SEQ) / 16, 13);
  highway_kernel<<<hgrid, 128, 0, stream>>>(XH0, XH1, WGb, WNb, WLb, hwbg, hwbn, hwbl);
  highway_kernel<<<hgrid, 128, 0, stream>>>(XH1, XH0,
      WGb + (size_t)EPAD * EPAD, WNb + (size_t)EPAD * EPAD, WLb + (size_t)EPAD * EPAD,
      hwbg + EIN, hwbn + EIN, hwbl + EIN);

  // --- persistent LSTM scan ---
  lstm_kernel<<<NBLK, 256, 0, stream>>>(XH0, WIHb, WHHb, bih, bhh, HBF, CST, GT, out, BAR);
}
